// MultiHeadAttention_7499012899130
// MI455X (gfx1250) — matrix operand order repaired and checked by reading, not compiled, not run on hardware
//
#include <hip/hip_runtime.h>
#include <hip/hip_bf16.h>

// ---------------- constants (match reference) ----------------
#define Bb 4
#define Ss 1024
#define Dd 1024
#define Hh 16
#define HD 64

typedef __bf16 bf16;
typedef __attribute__((ext_vector_type(4)))  bf16  v4bf;
typedef __attribute__((ext_vector_type(8)))  bf16  v8bf;
typedef __attribute__((ext_vector_type(16))) bf16  v16bf;
typedef __attribute__((ext_vector_type(4)))  float v4f;
typedef __attribute__((ext_vector_type(8)))  float v8f;

// ---------------- WMMA helpers ----------------
__device__ __forceinline__ v8f wmma_bf16(v16bf a, v16bf b, v8f c) {
  // D(16x16 f32) = A(16x32 bf16) * B(32x16 bf16) + C
  return __builtin_amdgcn_wmma_f32_16x16x32_bf16(
      /*neg_a=*/false, a, /*neg_b=*/false, b,
      /*c_mod=*/(short)0, c, /*reuse_a=*/false, /*reuse_b=*/false);
}

// A-fragment (16x32, 16-bit): lane l<16 -> row l, K = [0..7]+[16..23];
// lane l>=16 -> row l-16, K = [8..15]+[24..31]
__device__ __forceinline__ v16bf load_a_bf16(const bf16* A, int lda,
                                             int row0, int k0, int lane) {
  const bf16* p = A + (size_t)(row0 + (lane & 15)) * lda + k0 + ((lane >> 4) << 3);
  v8bf lo = *(const v8bf*)p;          // K offsets +0..7
  v8bf hi = *(const v8bf*)(p + 16);   // K offsets +16..23
  return __builtin_shufflevector(lo, hi, 0,1,2,3,4,5,6,7,8,9,10,11,12,13,14,15);
}

// Same A-fragment read out of an LDS-resident tile (row0 already removed)
__device__ __forceinline__ v16bf load_a_lds(const bf16* Atile, int lda,
                                            int k0, int lane) {
  const bf16* p = Atile + (size_t)(lane & 15) * lda + k0 + ((lane >> 4) << 3);
  v8bf lo = *(const v8bf*)p;
  v8bf hi = *(const v8bf*)(p + 16);
  return __builtin_shufflevector(lo, hi, 0,1,2,3,4,5,6,7,8,9,10,11,12,13,14,15);
}

// A-fragment from fp32 source (convert in-register, round-to-nearest)
__device__ __forceinline__ v16bf load_a_f32(const float* A, int lda,
                                            int row0, int k0, int lane) {
  const float* p = A + (size_t)(row0 + (lane & 15)) * lda + k0 + ((lane >> 4) << 3);
  v4f a0 = *(const v4f*)(p);
  v4f a1 = *(const v4f*)(p + 4);
  v4f b0 = *(const v4f*)(p + 16);
  v4f b1 = *(const v4f*)(p + 20);
  v16bf r;
#pragma unroll
  for (int i = 0; i < 4; ++i) {
    r[i]      = (bf16)a0[i];
    r[4 + i]  = (bf16)a1[i];
    r[8 + i]  = (bf16)b0[i];
    r[12 + i] = (bf16)b1[i];
  }
  return r;
}

// B-fragment (32x16, 16-bit), B stored transposed as Bt[N][K]:
// lane l<16 -> col l, K = [0..7]+[16..23] ; lane l>=16 -> col l-16, K = [8..15]+[24..31]
__device__ __forceinline__ v16bf load_b_bf16(const bf16* Bt, int ldb,
                                             int col0, int k0, int lane) {
  const bf16* p = Bt + (size_t)(col0 + (lane & 15)) * ldb + k0 + ((lane >> 4) << 3);
  v8bf lo = *(const v8bf*)p;
  v8bf hi = *(const v8bf*)(p + 16);
  return __builtin_shufflevector(lo, hi, 0,1,2,3,4,5,6,7,8,9,10,11,12,13,14,15);
}

// Async copy of one contiguous 32KB row-block (16 x 1024 bf16) into LDS.
// 256 threads x 8 x B128 = 2048 loads. ASYNCcnt path (CDNA5).
__device__ __forceinline__ void async_copy_rowblock(const bf16* __restrict__ src,
                                                    bf16* Atile, int t) {
#pragma unroll
  for (int i = 0; i < 8; ++i) {
    int idx = (t + i * 256) * 8;  // element index, 16 bytes per load
    unsigned lds = (unsigned)(uintptr_t)(&Atile[idx]);
    const bf16* g = src + idx;
    asm volatile("global_load_async_to_lds_b128 %0, %1, off"
                 :: "v"(lds), "v"(g) : "memory");
  }
  asm volatile("s_wait_asynccnt 0x0" ::: "memory");
}

// ---------------- conversion kernels ----------------
__global__ __launch_bounds__(256) void k_cvt_bf16(const float* __restrict__ x,
                                                  bf16* __restrict__ y, int n4) {
  int i = blockIdx.x * blockDim.x + threadIdx.x;
  if (i < n4) {
    v4f v = ((const v4f*)x)[i];
    v4bf o;
#pragma unroll
    for (int j = 0; j < 4; ++j) o[j] = (bf16)v[j];
    ((v4bf*)y)[i] = o;
  }
}

// Wt[c][r] = (bf16) W[r][c], 1024x1024, LDS-tiled transpose
__global__ __launch_bounds__(1024) void k_cvt_wT(const float* __restrict__ W,
                                                 bf16* __restrict__ Wt) {
  __shared__ float tile[32][33];
  int r0 = blockIdx.y * 32, c0 = blockIdx.x * 32;
  int tx = threadIdx.x, ty = threadIdx.y;
  tile[ty][tx] = W[(size_t)(r0 + ty) * Dd + c0 + tx];
  __syncthreads();
  Wt[(size_t)(c0 + ty) * Dd + r0 + tx] = (bf16)tile[tx][ty];
}

// ---------------- QKV projection ----------------
// C[M=4096,N=1024] = A(bf16[M,K=1024]) @ Bt(bf16[N,K])^T + bias
// All 8 waves of a block share one 16-row A block -> staged in LDS via async copy.
// K-loop: even/odd B buffers, 14 steady iterations (#pragma unroll 2), 2-step tail.
// mode 0: write head-split  out[((b*H+h)*S+s)*64 + hd]   (Q, K)
// mode 1: write transposed  out[((b*H+h)*64+hd)*S + s]   (V)
__global__ __launch_bounds__(256) void k_proj(const bf16* __restrict__ A,
                                              const bf16* __restrict__ Bt,
                                              const float* __restrict__ bias,
                                              bf16* __restrict__ out, int mode) {
  const int K = Dd;
  __shared__ __align__(16) bf16 Atile[16 * Dd];  // 32 KB
  int t = threadIdx.x;
  int row0 = (blockIdx.x >> 1) << 4;             // same for all 8 waves
  async_copy_rowblock(A + (size_t)row0 * K, Atile, t);
  __syncthreads();

  int wave = t >> 5, lane = t & 31;
  int col0 = ((blockIdx.x * 8 + wave) & 15) * 64;

  v8f acc[4] = {v8f{}, v8f{}, v8f{}, v8f{}};
  v16bf b0[4];
#pragma unroll
  for (int j = 0; j < 4; ++j) b0[j] = load_b_bf16(Bt, K, col0 + j * 16, 0, lane);

#pragma unroll 2
  for (int k0 = 0; k0 < K - 128; k0 += 64) {   // 14 iterations
    __builtin_prefetch(Bt + (size_t)(col0 + (lane & 15)) * K + k0 + 128, 0, 1);
    v16bf b1[4];
#pragma unroll
    for (int j = 0; j < 4; ++j) b1[j] = load_b_bf16(Bt, K, col0 + j * 16, k0 + 32, lane);
    v16bf a0 = load_a_lds(Atile, K, k0, lane);
#pragma unroll
    for (int j = 0; j < 4; ++j) acc[j] = wmma_bf16(a0, b0[j], acc[j]);
#pragma unroll
    for (int j = 0; j < 4; ++j) b0[j] = load_b_bf16(Bt, K, col0 + j * 16, k0 + 64, lane);
    v16bf a1 = load_a_lds(Atile, K, k0 + 32, lane);
#pragma unroll
    for (int j = 0; j < 4; ++j) acc[j] = wmma_bf16(a1, b1[j], acc[j]);
  }
  {  // tail step 1: k = K-128, K-96
    v16bf b1[4];
#pragma unroll
    for (int j = 0; j < 4; ++j) b1[j] = load_b_bf16(Bt, K, col0 + j * 16, K - 96, lane);
    v16bf a0 = load_a_lds(Atile, K, K - 128, lane);
#pragma unroll
    for (int j = 0; j < 4; ++j) acc[j] = wmma_bf16(a0, b0[j], acc[j]);
#pragma unroll
    for (int j = 0; j < 4; ++j) b0[j] = load_b_bf16(Bt, K, col0 + j * 16, K - 64, lane);
    v16bf a1 = load_a_lds(Atile, K, K - 96, lane);
#pragma unroll
    for (int j = 0; j < 4; ++j) acc[j] = wmma_bf16(a1, b1[j], acc[j]);
  }
  {  // tail step 2: k = K-64, K-32
    v16bf b1[4];
#pragma unroll
    for (int j = 0; j < 4; ++j) b1[j] = load_b_bf16(Bt, K, col0 + j * 16, K - 32, lane);
    v16bf a0 = load_a_lds(Atile, K, K - 64, lane);
#pragma unroll
    for (int j = 0; j < 4; ++j) acc[j] = wmma_bf16(a0, b0[j], acc[j]);
    v16bf a1 = load_a_lds(Atile, K, K - 32, lane);
#pragma unroll
    for (int j = 0; j < 4; ++j) acc[j] = wmma_bf16(a1, b1[j], acc[j]);
  }

  int nl = lane & 15, mb = (lane >> 4) << 3;
#pragma unroll
  for (int j = 0; j < 4; ++j) {
#pragma unroll
    for (int r = 0; r < 8; ++r) {
      int m = row0 + mb + r;
      int n = col0 + j * 16 + nl;
      float v = acc[j][r] + bias[n];
      int b = m >> 10, s = m & 1023, h = n >> 6, hd = n & 63;
      size_t o = (mode == 0)
                     ? (((size_t)(b * Hh + h) * Ss + s) * HD + hd)
                     : (((size_t)(b * Hh + h) * HD + hd) * Ss + s);
      out[o] = (bf16)v;
    }
  }
}

// ---------------- scores = (Q @ K^T) * 1/sqrt(64), fp32 into d_out ----------------
__global__ __launch_bounds__(256) void k_scores(const bf16* __restrict__ Qh,
                                                const bf16* __restrict__ Kh,
                                                float* __restrict__ scores) {
  int bh = blockIdx.y;
  const bf16* Aq = Qh + (size_t)bh * Ss * HD;
  const bf16* Bt = Kh + (size_t)bh * Ss * HD;   // K[b,h][s_k][hd] == Bt[N][K]
  float* Cs = scores + (size_t)bh * Ss * Ss;

  int wave = threadIdx.x >> 5, lane = threadIdx.x & 31;
  int tile = blockIdx.x * 8 + wave;             // 1024 tiles per bh
  int mt = tile >> 4, nt = tile & 15;
  int row0 = mt * 16, col0 = nt * 64;

  v8f acc[4] = {v8f{}, v8f{}, v8f{}, v8f{}};
#pragma unroll
  for (int k0 = 0; k0 < HD; k0 += 32) {
    v16bf a = load_a_bf16(Aq, HD, row0, k0, lane);
#pragma unroll
    for (int j = 0; j < 4; ++j) {
      v16bf b = load_b_bf16(Bt, HD, col0 + j * 16, k0, lane);
      acc[j] = wmma_bf16(a, b, acc[j]);
    }
  }
  const float scale = 0.125f;  // 1/sqrt(HD)
  int nl = lane & 15, mb = (lane >> 4) << 3;
#pragma unroll
  for (int j = 0; j < 4; ++j)
#pragma unroll
    for (int r = 0; r < 8; ++r)
      Cs[(size_t)(row0 + mb + r) * Ss + col0 + j * 16 + nl] = acc[j][r] * scale;
}

// ---------------- causal softmax, in place on fp32 weights ----------------
__global__ __launch_bounds__(256) void k_softmax(float* __restrict__ attnW) {
  __shared__ float red[256];
  int bh = blockIdx.y, srow = blockIdx.x, t = threadIdx.x;
  float* row = attnW + ((size_t)bh * Ss + srow) * Ss;

  v4f x = ((const v4f*)row)[t];
#pragma unroll
  for (int i = 0; i < 4; ++i)
    if (t * 4 + i > srow) x[i] -= 1e12f;  // additive causal mask (matches ref)

  float m = fmaxf(fmaxf(x[0], x[1]), fmaxf(x[2], x[3]));
  red[t] = m; __syncthreads();
  for (int s = 128; s > 0; s >>= 1) {
    if (t < s) red[t] = fmaxf(red[t], red[t + s]);
    __syncthreads();
  }
  float mx = red[0]; __syncthreads();

  v4f e;
  float psum = 0.f;
#pragma unroll
  for (int i = 0; i < 4; ++i) { e[i] = expf(x[i] - mx); psum += e[i]; }
  red[t] = psum; __syncthreads();
  for (int s = 128; s > 0; s >>= 1) {
    if (t < s) red[t] += red[t + s];
    __syncthreads();
  }
  float inv = 1.0f / red[0];
#pragma unroll
  for (int i = 0; i < 4; ++i) e[i] *= inv;
  ((v4f*)row)[t] = e;
}

// ---------------- ctx = P @ V  (A = fp32 weights converted on the fly) ----------------
__global__ __launch_bounds__(256) void k_attnv(const float* __restrict__ attnW,
                                               const bf16* __restrict__ Vt,
                                               bf16* __restrict__ ctx) {
  int bh = blockIdx.y;
  const float* Ap = attnW + (size_t)bh * Ss * Ss;
  const bf16* Bt = Vt + (size_t)bh * HD * Ss;    // Vt[b,h][hd][s_k] == Bt[N=64][K=S]
  int b = bh / Hh, h = bh % Hh;

  int wave = threadIdx.x >> 5, lane = threadIdx.x & 31;
  int mt = blockIdx.x * 8 + wave;                // 64 row-tiles per bh
  int row0 = mt * 16;
  const int K = Ss;

  v8f acc[4] = {v8f{}, v8f{}, v8f{}, v8f{}};
  v16bf b0[4];
#pragma unroll
  for (int j = 0; j < 4; ++j) b0[j] = load_b_bf16(Bt, K, j * 16, 0, lane);

#pragma unroll 2
  for (int k0 = 0; k0 < K - 128; k0 += 64) {   // 14 iterations
    __builtin_prefetch(Ap + (size_t)(row0 + (lane & 15)) * K + k0 + 64, 0, 1);
    v16bf b1[4];
#pragma unroll
    for (int j = 0; j < 4; ++j) b1[j] = load_b_bf16(Bt, K, j * 16, k0 + 32, lane);
    v16bf a0 = load_a_f32(Ap, K, row0, k0, lane);
#pragma unroll
    for (int j = 0; j < 4; ++j) acc[j] = wmma_bf16(a0, b0[j], acc[j]);
#pragma unroll
    for (int j = 0; j < 4; ++j) b0[j] = load_b_bf16(Bt, K, j * 16, k0 + 64, lane);
    v16bf a1 = load_a_f32(Ap, K, row0, k0 + 32, lane);
#pragma unroll
    for (int j = 0; j < 4; ++j) acc[j] = wmma_bf16(a1, b1[j], acc[j]);
  }
  {  // tail step 1
    v16bf b1[4];
#pragma unroll
    for (int j = 0; j < 4; ++j) b1[j] = load_b_bf16(Bt, K, j * 16, K - 96, lane);
    v16bf a0 = load_a_f32(Ap, K, row0, K - 128, lane);
#pragma unroll
    for (int j = 0; j < 4; ++j) acc[j] = wmma_bf16(a0, b0[j], acc[j]);
#pragma unroll
    for (int j = 0; j < 4; ++j) b0[j] = load_b_bf16(Bt, K, j * 16, K - 64, lane);
    v16bf a1 = load_a_f32(Ap, K, row0, K - 96, lane);
#pragma unroll
    for (int j = 0; j < 4; ++j) acc[j] = wmma_bf16(a1, b1[j], acc[j]);
  }
  {  // tail step 2
    v16bf b1[4];
#pragma unroll
    for (int j = 0; j < 4; ++j) b1[j] = load_b_bf16(Bt, K, j * 16, K - 32, lane);
    v16bf a0 = load_a_f32(Ap, K, row0, K - 64, lane);
#pragma unroll
    for (int j = 0; j < 4; ++j) acc[j] = wmma_bf16(a0, b0[j], acc[j]);
    v16bf a1 = load_a_f32(Ap, K, row0, K - 32, lane);
#pragma unroll
    for (int j = 0; j < 4; ++j) acc[j] = wmma_bf16(a1, b1[j], acc[j]);
  }

  int nl = lane & 15, mb = (lane >> 4) << 3;
#pragma unroll
  for (int j = 0; j < 4; ++j)
#pragma unroll
    for (int r = 0; r < 8; ++r) {
      int s = row0 + mb + r;
      int hd = j * 16 + nl;
      ctx[((size_t)(b * Ss + s)) * Dd + h * HD + hd] = (bf16)acc[j][r];
    }
}

// ---------------- out = ctx @ Wo + bo, fp32 into d_out ----------------
__global__ __launch_bounds__(256) void k_outproj(const bf16* __restrict__ A,
                                                 const bf16* __restrict__ Bt,
                                                 const float* __restrict__ bias,
                                                 float* __restrict__ out) {
  const int K = Dd;
  __shared__ __align__(16) bf16 Atile[16 * Dd];  // 32 KB
  int t = threadIdx.x;
  int row0 = (blockIdx.x >> 1) << 4;
  async_copy_rowblock(A + (size_t)row0 * K, Atile, t);
  __syncthreads();

  int wave = t >> 5, lane = t & 31;
  int col0 = ((blockIdx.x * 8 + wave) & 15) * 64;

  v8f acc[4] = {v8f{}, v8f{}, v8f{}, v8f{}};
  v16bf b0[4];
#pragma unroll
  for (int j = 0; j < 4; ++j) b0[j] = load_b_bf16(Bt, K, col0 + j * 16, 0, lane);

#pragma unroll 2
  for (int k0 = 0; k0 < K - 128; k0 += 64) {   // 14 iterations
    __builtin_prefetch(Bt + (size_t)(col0 + (lane & 15)) * K + k0 + 128, 0, 1);
    v16bf b1[4];
#pragma unroll
    for (int j = 0; j < 4; ++j) b1[j] = load_b_bf16(Bt, K, col0 + j * 16, k0 + 32, lane);
    v16bf a0 = load_a_lds(Atile, K, k0, lane);
#pragma unroll
    for (int j = 0; j < 4; ++j) acc[j] = wmma_bf16(a0, b0[j], acc[j]);
#pragma unroll
    for (int j = 0; j < 4; ++j) b0[j] = load_b_bf16(Bt, K, col0 + j * 16, k0 + 64, lane);
    v16bf a1 = load_a_lds(Atile, K, k0 + 32, lane);
#pragma unroll
    for (int j = 0; j < 4; ++j) acc[j] = wmma_bf16(a1, b1[j], acc[j]);
  }
  {  // tail step 1
    v16bf b1[4];
#pragma unroll
    for (int j = 0; j < 4; ++j) b1[j] = load_b_bf16(Bt, K, col0 + j * 16, K - 96, lane);
    v16bf a0 = load_a_lds(Atile, K, K - 128, lane);
#pragma unroll
    for (int j = 0; j < 4; ++j) acc[j] = wmma_bf16(a0, b0[j], acc[j]);
#pragma unroll
    for (int j = 0; j < 4; ++j) b0[j] = load_b_bf16(Bt, K, col0 + j * 16, K - 64, lane);
    v16bf a1 = load_a_lds(Atile, K, K - 96, lane);
#pragma unroll
    for (int j = 0; j < 4; ++j) acc[j] = wmma_bf16(a1, b1[j], acc[j]);
  }
  {  // tail step 2
    v16bf b1[4];
#pragma unroll
    for (int j = 0; j < 4; ++j) b1[j] = load_b_bf16(Bt, K, col0 + j * 16, K - 32, lane);
    v16bf a0 = load_a_lds(Atile, K, K - 64, lane);
#pragma unroll
    for (int j = 0; j < 4; ++j) acc[j] = wmma_bf16(a0, b0[j], acc[j]);
    v16bf a1 = load_a_lds(Atile, K, K - 32, lane);
#pragma unroll
    for (int j = 0; j < 4; ++j) acc[j] = wmma_bf16(a1, b1[j], acc[j]);
  }

  int nl = lane & 15, mb = (lane >> 4) << 3;
#pragma unroll
  for (int j = 0; j < 4; ++j)
#pragma unroll
    for (int r = 0; r < 8; ++r) {
      int n = col0 + j * 16 + nl;
      out[(size_t)(row0 + mb + r) * Dd + n] = acc[j][r] + bias[n];
    }
}

// ---------------- host launcher ----------------
extern "C" void kernel_launch(void* const* d_in, const int* in_sizes, int n_in,
                              void* d_out, int out_size, void* d_ws, size_t ws_size,
                              hipStream_t stream) {
  const float* q    = (const float*)d_in[0];
  const float* k    = (const float*)d_in[1];
  const float* v    = (const float*)d_in[2];
  // d_in[3] = mask (causal, applied analytically)
  const float* Wq   = (const float*)d_in[4];
  const float* bq   = (const float*)d_in[5];
  const float* Wk   = (const float*)d_in[6];
  const float* bk   = (const float*)d_in[7];
  const float* Wv   = (const float*)d_in[8];
  const float* bv   = (const float*)d_in[9];
  const float* Wo   = (const float*)d_in[10];
  const float* bo   = (const float*)d_in[11];

  float* out   = (float*)d_out;                      // [B,S,D]
  float* attnW = out + (size_t)Bb * Ss * Dd;         // [B,H,S,S]

  char* w = (char*)d_ws;
  const size_t MB = 1u << 20;
  bf16* qb  = (bf16*)(w + 0 * MB);    // [B*S, D]
  bf16* kb  = (bf16*)(w + 8 * MB);
  bf16* vb  = (bf16*)(w + 16 * MB);
  bf16* WqT = (bf16*)(w + 24 * MB);   // [N,K] each 2MB
  bf16* WkT = (bf16*)(w + 26 * MB);
  bf16* WvT = (bf16*)(w + 28 * MB);
  bf16* WoT = (bf16*)(w + 30 * MB);
  bf16* Qh  = (bf16*)(w + 32 * MB);   // [B,H,S,64]
  bf16* Kh  = (bf16*)(w + 40 * MB);   // [B,H,S,64]
  bf16* Vt  = (bf16*)(w + 48 * MB);   // [B,H,64,S]
  bf16* ctx = (bf16*)(w + 56 * MB);   // [B,S,D]

  const int n4 = (Bb * Ss * Dd) / 4;
  dim3 blk(256);
  // 1) convert inputs
  k_cvt_bf16<<<dim3((n4 + 255) / 256), blk, 0, stream>>>(q, qb, n4);
  k_cvt_bf16<<<dim3((n4 + 255) / 256), blk, 0, stream>>>(k, kb, n4);
  k_cvt_bf16<<<dim3((n4 + 255) / 256), blk, 0, stream>>>(v, vb, n4);
  // 2) transpose+convert weights
  dim3 tgrid(Dd / 32, Dd / 32), tblk(32, 32);
  k_cvt_wT<<<tgrid, tblk, 0, stream>>>(Wq, WqT);
  k_cvt_wT<<<tgrid, tblk, 0, stream>>>(Wk, WkT);
  k_cvt_wT<<<tgrid, tblk, 0, stream>>>(Wv, WvT);
  k_cvt_wT<<<tgrid, tblk, 0, stream>>>(Wo, WoT);
  // 3) QKV projections (512 blocks x 8 waves = 4096 wave-tiles of 16x64)
  k_proj<<<dim3(512), blk, 0, stream>>>(qb, WqT, bq, Qh, 0);
  k_proj<<<dim3(512), blk, 0, stream>>>(kb, WkT, bk, Kh, 0);
  k_proj<<<dim3(512), blk, 0, stream>>>(vb, WvT, bv, Vt, 1);
  // 4) scores -> fp32 attention region of d_out
  k_scores<<<dim3(128, Bb * Hh), blk, 0, stream>>>(Qh, Kh, attnW);
  // 5) causal softmax in place
  k_softmax<<<dim3(Ss, Bb * Hh), blk, 0, stream>>>(attnW);
  // 6) ctx = P @ V
  k_attnv<<<dim3(8, Bb * Hh), blk, 0, stream>>>(attnW, Vt, ctx);
  // 7) out = ctx @ Wo + bo
  k_outproj<<<dim3(512), blk, 0, stream>>>(ctx, WoT, bo, out);
}
